// DynamicConv2d_47485158425360
// MI455X (gfx1250) — compile-verified, matrix operand order checked by reading, not run on hardware
//
#include <hip/hip_runtime.h>

typedef __attribute__((ext_vector_type(16))) __bf16 v16bf;
typedef __attribute__((ext_vector_type(8)))  __bf16 v8bf;
typedef __attribute__((ext_vector_type(8)))  float  v8f;

#define CIN   128
#define COUT  256
#define KEXP  8
#define HID   32
#define BATCH 32
#define HH    56
#define WW    56
#define HW    (HH*WW)      // 3136
#define HP    58           // padded H
#define WP    58           // padded W
#define TILE_M 128
#define TILE_N 128
#define LDT    40          // LDS row stride in bf16 (80B, 16B-aligned rows/chunks)
#define NSTEP  36          // 9 kernel taps * 4 cin-slices of 32

// ---------------- Kernel 1: routing (GAP -> fc1 -> relu -> fc2 -> softmax) ----
__global__ void routing_kernel(const float* __restrict__ x,
                               const float* __restrict__ fc1_w,
                               const float* __restrict__ fc1_b,
                               const float* __restrict__ fc2_w,
                               const float* __restrict__ fc2_b,
                               float* __restrict__ a_out) {
    __shared__ float partial[256];
    __shared__ float vsh[CIN];
    __shared__ float hsh[HID];
    __shared__ float lsh[KEXP];
    const int b   = blockIdx.x;
    const int tid = threadIdx.x;
    const int c    = tid & 127;
    const int half = tid >> 7;
    const float* xp = x + ((size_t)b * CIN + c) * HW + half * (HW / 2);
    float s = 0.f;
    for (int i = 0; i < HW / 2; ++i) s += xp[i];
    partial[tid] = s;
    __syncthreads();
    if (tid < CIN) vsh[tid] = (partial[tid] + partial[tid + 128]) * (1.0f / (float)HW);
    __syncthreads();
    if (tid < HID) {
        float h = fc1_b[tid];
        const float* wp = fc1_w + tid * CIN;
        for (int cc = 0; cc < CIN; ++cc) h += vsh[cc] * wp[cc];
        hsh[tid] = fmaxf(h, 0.f);
    }
    __syncthreads();
    if (tid < KEXP) {
        float l = fc2_b[tid];
        const float* wp = fc2_w + tid * HID;
        for (int j = 0; j < HID; ++j) l += hsh[j] * wp[j];
        lsh[tid] = l;
    }
    __syncthreads();
    if (tid == 0) {
        float mx = lsh[0];
        for (int k = 1; k < KEXP; ++k) mx = fmaxf(mx, lsh[k]);
        float e[KEXP]; float sum = 0.f;
        for (int k = 0; k < KEXP; ++k) { e[k] = __expf(lsh[k] - mx); sum += e[k]; }
        float inv = 1.f / sum;
        for (int k = 0; k < KEXP; ++k) a_out[b * KEXP + k] = e[k] * inv;
    }
}

// ---------------- Kernel 2: mix bank -> per-sample bf16 weights ---------------
// wdyn layout: [b][pos(=kh*3+kw)][cout][cin]  (cin contiguous = GEMM-K contiguous)
__global__ void wdyn_kernel(const float* __restrict__ bank,
                            const float* __restrict__ a,
                            __bf16* __restrict__ wdyn) {
    const int b    = blockIdx.y;
    const int cout = blockIdx.x;
    const int cin  = threadIdx.x;          // 128 threads
    __shared__ float ash[KEXP];
    if (threadIdx.x < KEXP) ash[threadIdx.x] = a[b * KEXP + threadIdx.x];
    __syncthreads();
    float acc[9];
#pragma unroll
    for (int p = 0; p < 9; ++p) acc[p] = 0.f;
    for (int k = 0; k < KEXP; ++k) {
        const float ak = ash[k];
        const float* wp = bank + (((size_t)k * COUT + cout) * CIN + cin) * 9;
#pragma unroll
        for (int p = 0; p < 9; ++p) acc[p] += ak * wp[p];
    }
#pragma unroll
    for (int p = 0; p < 9; ++p)
        wdyn[(((size_t)b * 9 + p) * COUT + cout) * CIN + cin] = (__bf16)acc[p];
}

// ---------------- Kernel 3: NCHW f32 -> padded NHWC bf16 ----------------------
// xpad layout: [b][ih(0..57)][iw(0..57)][cin], zero halo border
__global__ void xpad_kernel(const float* __restrict__ x,
                            __bf16* __restrict__ xpad) {
    const int b  = blockIdx.y;
    const int ih = blockIdx.x;             // 0..57
    const int c  = threadIdx.x & 127;
    const int sh = threadIdx.x >> 7;       // 0..1
    const bool hok = (ih >= 1) && (ih <= HH);
    const float* xb = x + ((size_t)b * CIN + c) * HW + (ih - 1) * WW - 1;
    __bf16* ob = xpad + (((size_t)b * HP + ih) * WP) * CIN + c;
    for (int j = 0; j < WP / 2; ++j) {
        const int iw = 2 * j + sh;
        const bool ok = hok && (iw >= 1) && (iw <= WW);
        const float v = ok ? xb[iw] : 0.f;
        ob[(size_t)iw * CIN] = (__bf16)v;
    }
}

// ---------------- async global->LDS staging (2 x 16B chunks per operand) ------
__device__ __forceinline__ void async_stage(unsigned ldsA, unsigned ldsB,
                                            unsigned offA, unsigned offB,
                                            const __bf16* wbase,
                                            const __bf16* xbase) {
    asm volatile("global_load_async_to_lds_b128 %0, %1, %2 offset:0\n\t"
                 "global_load_async_to_lds_b128 %0, %1, %2 offset:16"
                 :: "v"(ldsA), "v"(offA), "s"(wbase) : "memory");
    asm volatile("global_load_async_to_lds_b128 %0, %1, %2 offset:0\n\t"
                 "global_load_async_to_lds_b128 %0, %1, %2 offset:16"
                 :: "v"(ldsB), "v"(offB), "s"(xbase) : "memory");
}

// ---------------- Kernel 4: implicit-GEMM conv via bf16 WMMA ------------------
__global__ void __launch_bounds__(256)
conv_wmma_kernel(const __bf16* __restrict__ xpad,
                 const __bf16* __restrict__ wdyn,
                 float* __restrict__ y) {
    // double-buffered tiles: [buf][row 0..127][k 0..31], row stride LDT
    __shared__ __attribute__((aligned(16))) __bf16 As[2 * TILE_M * LDT];
    __shared__ __attribute__((aligned(16))) __bf16 Bs[2 * TILE_N * LDT];

    const int tid  = threadIdx.x;
    const int lane = tid & 31;
    const int wave = tid >> 5;
    const int waveM = wave & 1;   // 0..1  -> 64 rows each
    const int waveN = wave >> 1;  // 0..3  -> 32 cols each
    const int lrow  = lane & 15;
    const int khalf = lane >> 4;

    const int b        = blockIdx.z;
    const int coutBase = blockIdx.y * TILE_M;
    const int pBase    = blockIdx.x * TILE_N;

    // staging thread mapping: sm = tile row/col (0..127), sh = k-half (0..1)
    const int sm = tid & 127;
    const int sh = tid >> 7;

    const __bf16* Wb = wdyn + (size_t)b * 9 * COUT * CIN;
    const __bf16* Xb = xpad + (size_t)b * HP * WP * CIN;

    // output column for B staging (clamped for the ragged last tile; those
    // columns compute garbage but are never stored)
    const int p  = pBase + sm;
    const int pc = (p < HW) ? p : (HW - 1);
    const int oh = pc / WW;
    const int ow = pc - oh * WW;

    // per-thread byte-offset bases (k-contiguous layouts)
    const unsigned constA = (unsigned)(((coutBase + sm) * CIN + sh * 16) * 2);
    const unsigned constB = (unsigned)(((oh * WP + ow) * CIN + sh * 16) * 2);

    // LDS destination addresses for this thread's two 16B chunks (offset:0/16)
    const unsigned ldsA0 = (unsigned)(size_t)&As[sm * LDT + sh * 16];
    const unsigned ldsA1 = (unsigned)(size_t)&As[TILE_M * LDT + sm * LDT + sh * 16];
    const unsigned ldsB0 = (unsigned)(size_t)&Bs[sm * LDT + sh * 16];
    const unsigned ldsB1 = (unsigned)(size_t)&Bs[TILE_N * LDT + sm * LDT + sh * 16];

    v8f acc[4][2];
#pragma unroll
    for (int mi = 0; mi < 4; ++mi)
#pragma unroll
        for (int ni = 0; ni < 2; ++ni) acc[mi][ni] = (v8f)(0.f);

    auto issue = [&](int s) {
        const int posn = s >> 2;            // 0..8 tap
        const int csn  = s & 3;             // cin slice
        const int khn  = (posn * 11) >> 5;  // posn/3 for posn<9
        const int kwn  = posn - khn * 3;
        const unsigned offA = constA + (unsigned)posn * (COUT * CIN * 2) + csn * 64;
        const unsigned offB = constB + (unsigned)khn * (WP * CIN * 2)
                                     + (unsigned)kwn * (CIN * 2) + csn * 64;
        async_stage((s & 1) ? ldsA1 : ldsA0, (s & 1) ? ldsB1 : ldsB0,
                    offA, offB, Wb, Xb);
    };

    issue(0);

    for (int step = 0; step < NSTEP; ++step) {
        if (step + 1 < NSTEP) {
            issue(step + 1);
            asm volatile("s_wait_asynccnt 4" ::: "memory");   // step's 4 copies done
        } else {
            asm volatile("s_wait_asynccnt 0" ::: "memory");
        }
        __syncthreads();                                      // whole tile visible

        const __bf16* Abuf = &As[(step & 1) * (TILE_M * LDT)];
        const __bf16* Bbuf = &Bs[(step & 1) * (TILE_N * LDT)];

        // fragment loads: 2 x ds_load_b128 per fragment (bf16 A/B VGPR layout)
        v16bf afr[4];
#pragma unroll
        for (int mi = 0; mi < 4; ++mi) {
            const int m = waveM * 64 + mi * 16 + lrow;
            ((v8bf*)&afr[mi])[0] = *(const v8bf*)&Abuf[m * LDT + khalf * 8];
            ((v8bf*)&afr[mi])[1] = *(const v8bf*)&Abuf[m * LDT + 16 + khalf * 8];
        }
        v16bf bfr[2];
#pragma unroll
        for (int ni = 0; ni < 2; ++ni) {
            const int n = waveN * 32 + ni * 16 + lrow;
            ((v8bf*)&bfr[ni])[0] = *(const v8bf*)&Bbuf[n * LDT + khalf * 8];
            ((v8bf*)&bfr[ni])[1] = *(const v8bf*)&Bbuf[n * LDT + 16 + khalf * 8];
        }
#pragma unroll
        for (int mi = 0; mi < 4; ++mi)
#pragma unroll
            for (int ni = 0; ni < 2; ++ni)
                acc[mi][ni] = __builtin_amdgcn_wmma_f32_16x16x32_bf16(
                    false, afr[mi], false, bfr[ni],
                    (short)0, acc[mi][ni], false, false);

        __syncthreads();   // all waves done reading buf before it is refilled
    }

    // epilogue: C layout -> lane = column n, vgpr v + 8*(lane>>4) = row m
    float* yb = y + (size_t)b * COUT * HW;
#pragma unroll
    for (int mi = 0; mi < 4; ++mi) {
#pragma unroll
        for (int ni = 0; ni < 2; ++ni) {
            const int pn = pBase + waveN * 32 + ni * 16 + lrow;
            if (pn < HW) {
                const int m0 = coutBase + waveM * 64 + mi * 16 + khalf * 8;
#pragma unroll
                for (int v = 0; v < 8; ++v)
                    yb[(size_t)(m0 + v) * HW + pn] = acc[mi][ni][v];
            }
        }
    }
}

// ---------------- launcher ----------------------------------------------------
extern "C" void kernel_launch(void* const* d_in, const int* in_sizes, int n_in,
                              void* d_out, int out_size, void* d_ws, size_t ws_size,
                              hipStream_t stream) {
    const float* x     = (const float*)d_in[0];
    const float* bank  = (const float*)d_in[1];
    const float* fc1_w = (const float*)d_in[2];
    const float* fc1_b = (const float*)d_in[3];
    const float* fc2_w = (const float*)d_in[4];
    const float* fc2_b = (const float*)d_in[5];
    float* y = (float*)d_out;

    float*  a_ws = (float*)d_ws;
    __bf16* wdyn = (__bf16*)((char*)d_ws + 1024);
    size_t  wdynBytes = (size_t)BATCH * 9 * COUT * CIN * sizeof(__bf16); // 18.9 MB
    __bf16* xpad = (__bf16*)((char*)d_ws + 1024 + wdynBytes);            // 27.6 MB

    routing_kernel<<<BATCH, 256, 0, stream>>>(x, fc1_w, fc1_b, fc2_w, fc2_b, a_ws);
    wdyn_kernel<<<dim3(COUT, BATCH), 128, 0, stream>>>(bank, a_ws, wdyn);
    xpad_kernel<<<dim3(HP, BATCH), 256, 0, stream>>>(x, xpad);
    conv_wmma_kernel<<<dim3((HW + TILE_N - 1) / TILE_N, COUT / TILE_M, BATCH),
                       256, 0, stream>>>(xpad, wdyn, y);
}
